// Net_22565758173650
// MI455X (gfx1250) — matrix operand order repaired and checked by reading, not compiled, not run on hardware
//
#include <hip/hip_runtime.h>
#include <stdint.h>

#define VOCAB   50257
#define HID     768
#define BATCH   64
#define MAXLEN  32
#define SOS_IDX 1

typedef _Float16 v16h __attribute__((ext_vector_type(16)));
typedef _Float16 v8h  __attribute__((ext_vector_type(8)));
typedef _Float16 v4h  __attribute__((ext_vector_type(4)));
typedef float    v8f  __attribute__((ext_vector_type(8)));
typedef float    f8v  __attribute__((ext_vector_type(8)));
typedef float    f4v  __attribute__((ext_vector_type(4)));

static __device__ __forceinline__ float sigmoid_(float x) {
  return 1.0f / (1.0f + __expf(-x));
}

// Monotonic key: larger float -> larger unsigned key (handles sign bit).
static __device__ __forceinline__ unsigned int order_key(float x) {
  unsigned int b = __float_as_uint(x);
  return (b & 0x80000000u) ? ~b : (b | 0x80000000u);
}

static __device__ __forceinline__ v16h cat16(v8h lo, v8h hi) {
  return __builtin_shufflevector(lo, hi, 0, 1, 2, 3, 4, 5, 6, 7,
                                         8, 9, 10, 11, 12, 13, 14, 15);
}

// Decode a packed argmax entry back to a token index.
static __device__ __forceinline__ int decode_tok(unsigned long long p) {
  unsigned int n = ~(unsigned int)(p & 0xFFFFFFFFull);
  return (p != 0ull && n < VOCAB) ? (int)n : 0;
}

// ---------------------------------------------------------------- fp32->fp16
// One-time pass; source reads are non-temporal (never needed again).
__global__ void cvt_f32_to_f16(const float* __restrict__ in,
                               _Float16* __restrict__ out, int n4) {
  int i = blockIdx.x * blockDim.x + threadIdx.x;
  int stride = gridDim.x * blockDim.x;
  for (; i < n4; i += stride) {
    const f4v* src = reinterpret_cast<const f4v*>(in + 4 * (size_t)i);
    f4v x = __builtin_nontemporal_load(src);
    v4h y = __builtin_convertvector(x, v4h);
    *reinterpret_cast<v4h*>(out + 4 * (size_t)i) = y;  // RT: keep f16 in L2
  }
}

// ---------------------------------------------------------------- init state
__global__ void init_state(float* __restrict__ c_state,
                           _Float16* __restrict__ h16,
                           unsigned long long* __restrict__ arg_prev) {
  int i = blockIdx.x * blockDim.x + threadIdx.x;
  if (i < BATCH * HID) {
    c_state[i] = 0.0f;
    h16[i] = (_Float16)0.0f;
  }
  if (i < BATCH) {
    // Seed "previous argmax" with the SOS token for step 0.
    arg_prev[i] =
        (1ull << 32) | (unsigned long long)(~(unsigned int)SOS_IDX);
  }
}

// ------------------------------------------------ gates GEMM + LSTM cell
// grid = (6, 4), block = 256 (8 waves).
// Wave computes a 16(batch) x 16(j) tile of all four gates (i,f,g,o) via WMMA,
// then performs the cell update in registers. The previous step's token is
// decoded inline from arg_prev (no separate finalize kernel).
__launch_bounds__(256)
__global__ void gates_kernel(const float* __restrict__ emb,
                             const _Float16* __restrict__ Wih,
                             const _Float16* __restrict__ Whh,
                             const float* __restrict__ b_ih,
                             const float* __restrict__ b_hh,
                             const unsigned long long* __restrict__ arg_prev,
                             unsigned long long* __restrict__ arg_cur,
                             const _Float16* __restrict__ h16_in,
                             float* __restrict__ c_state,
                             float* __restrict__ h_f32,
                             _Float16* __restrict__ h16_out) {
  const int lane = threadIdx.x & 31;
  const int wave = threadIdx.x >> 5;
  const int m0 = blockIdx.y * 16;
  const int j0 = (blockIdx.x * 8 + wave) * 16;
  const int rlan = lane & 15;
  const int kbA = (lane < 16) ? 0 : 16;   // A-matrix per-lane K base
  const int kbB = (lane < 16) ? 0 : 16;   // B-matrix per-lane K base

  // Reset this step's argmax accumulator (only this kernel touches arg_cur;
  // the logits kernel of this step runs strictly after on the stream).
  if (blockIdx.x == 0 && blockIdx.y == 0 && threadIdx.x < BATCH)
    arg_cur[threadIdx.x] = 0ull;

  const int arow = m0 + rlan;
  const int tk = decode_tok(arg_prev[arow]);
  const float* xRow = emb + (size_t)tk * HID;             // embedding gather
  const _Float16* hRow = h16_in + (size_t)arow * HID;

  v8f acc[4] = {};  // i, f, g, o accumulators (same 16x16 position each)

  // x @ W_ih^T  (A converted f32->f16 on the fly; only 64 rows used)
  for (int k0 = 0; k0 < HID; k0 += 32) {
    f8v xlo = *reinterpret_cast<const f8v*>(xRow + k0 + kbA);
    f8v xhi = *reinterpret_cast<const f8v*>(xRow + k0 + kbA + 8);
    v16h A = cat16(__builtin_convertvector(xlo, v8h),
                   __builtin_convertvector(xhi, v8h));
#pragma unroll
    for (int g = 0; g < 4; ++g) {
      const _Float16* bp = Wih + (size_t)(j0 + g * HID + rlan) * HID + k0 + kbB;
      v16h Bm = *reinterpret_cast<const v16h*>(bp);
      acc[g] = __builtin_amdgcn_wmma_f32_16x16x32_f16(
          false, A, false, Bm, (short)0, acc[g], false, false);
    }
  }
  // h @ W_hh^T
  for (int k0 = 0; k0 < HID; k0 += 32) {
    v8h alo = *reinterpret_cast<const v8h*>(hRow + k0 + kbA);
    v8h ahi = *reinterpret_cast<const v8h*>(hRow + k0 + kbA + 8);
    v16h A = cat16(alo, ahi);
#pragma unroll
    for (int g = 0; g < 4; ++g) {
      const _Float16* bp = Whh + (size_t)(j0 + g * HID + rlan) * HID + k0 + kbB;
      v16h Bm = *reinterpret_cast<const v16h*>(bp);
      acc[g] = __builtin_amdgcn_wmma_f32_16x16x32_f16(
          false, A, false, Bm, (short)0, acc[g], false, false);
    }
  }

  // Cell update. Lane holds column j = j0 + rlan for 8 batch rows.
  const int j = j0 + rlan;
  const float bi = b_ih[j] + b_hh[j];
  const float bf = b_ih[j + HID] + b_hh[j + HID];
  const float bg = b_ih[j + 2 * HID] + b_hh[j + 2 * HID];
  const float bo = b_ih[j + 3 * HID] + b_hh[j + 3 * HID];
  const int roff = (lane < 16) ? 0 : 8;
#pragma unroll
  for (int v = 0; v < 8; ++v) {
    const int br = m0 + v + roff;
    const size_t idx = (size_t)br * HID + j;
    float iv = sigmoid_(acc[0][v] + bi);
    float fv = sigmoid_(acc[1][v] + bf);
    float gv = tanhf(acc[2][v] + bg);
    float ov = sigmoid_(acc[3][v] + bo);
    float cn = fv * c_state[idx] + iv * gv;
    float hn = ov * tanhf(cn);
    c_state[idx] = cn;
    h_f32[idx] = hn;                 // final-step value is the returned h
    h16_out[idx] = (_Float16)hn;     // f16 copy feeds the GEMMs
  }
}

// ------------------------------------------------ logits GEMM + argmax
// grid = (99, 2), block = 256 (8 waves).
// Wave: 32 batch rows (2 m-tiles) x 64 vocab cols (4 n-tiles). Sharing each
// B fragment across 2 WMMAs halves the per-step L2 read traffic of the
// L2-resident f16 W_out (154MB/step instead of 308MB/step).
// Logits are streamed out with non-temporal stores so the 412MB output stream
// cannot evict W_out from L2.
__launch_bounds__(256)
__global__ void logits_kernel(const _Float16* __restrict__ h16,
                              const _Float16* __restrict__ Wout,
                              const float* __restrict__ b_out,
                              float* __restrict__ logits,  // [B][MAXLEN][VOCAB]
                              int t,
                              unsigned long long* __restrict__ argbuf) {
  __shared__ unsigned long long sbest[32];
  const int lane = threadIdx.x & 31;
  const int wave = threadIdx.x >> 5;
  const int m0 = blockIdx.y * 32;                 // rows m0 .. m0+31
  const int n0 = (blockIdx.x * 8 + wave) * 64;
  const int rlan = lane & 15;
  const int kbA = (lane < 16) ? 0 : 16;
  const int kbB = (lane < 16) ? 0 : 16;

  if (threadIdx.x < 32) sbest[threadIdx.x] = 0ull;
  __syncthreads();

  const _Float16* aRow0 = h16 + (size_t)(m0 + rlan) * HID;
  const _Float16* aRow1 = h16 + (size_t)(m0 + 16 + rlan) * HID;

  v8f acc[2][4] = {};
  for (int k0 = 0; k0 < HID; k0 += 32) {
    v16h A0 = cat16(*reinterpret_cast<const v8h*>(aRow0 + k0 + kbA),
                    *reinterpret_cast<const v8h*>(aRow0 + k0 + kbA + 8));
    v16h A1 = cat16(*reinterpret_cast<const v8h*>(aRow1 + k0 + kbA),
                    *reinterpret_cast<const v8h*>(aRow1 + k0 + kbA + 8));
#pragma unroll
    for (int tI = 0; tI < 4; ++tI) {
      int nr = n0 + tI * 16 + rlan;
      nr = (nr < VOCAB) ? nr : (VOCAB - 1);  // clamp; OOB never stored/argmaxed
      const _Float16* bp = Wout + (size_t)nr * HID + k0 + kbB;
      v16h Bm = *reinterpret_cast<const v16h*>(bp);
      acc[0][tI] = __builtin_amdgcn_wmma_f32_16x16x32_f16(
          false, A0, false, Bm, (short)0, acc[0][tI], false, false);
      acc[1][tI] = __builtin_amdgcn_wmma_f32_16x16x32_f16(
          false, A1, false, Bm, (short)0, acc[1][tI], false, false);
    }
  }

  const int roff = (lane < 16) ? 0 : 8;
  unsigned long long best[2][8];
#pragma unroll
  for (int mt = 0; mt < 2; ++mt)
#pragma unroll
    for (int v = 0; v < 8; ++v) best[mt][v] = 0ull;

#pragma unroll
  for (int tI = 0; tI < 4; ++tI) {
    const int n = n0 + tI * 16 + rlan;
    const bool ok = (n < VOCAB);
    const float bias = ok ? b_out[n] : 0.0f;
    const unsigned long long nkey =
        (unsigned long long)(~(unsigned int)n);  // ~n: ties -> lowest index
#pragma unroll
    for (int mt = 0; mt < 2; ++mt) {
#pragma unroll
      for (int v = 0; v < 8; ++v) {
        float val = acc[mt][tI][v] + bias;
        if (ok) {
          const int br = m0 + mt * 16 + v + roff;
          float* dst = logits + (size_t)br * (MAXLEN * (size_t)VOCAB) +
                       (size_t)t * VOCAB + n;
          __builtin_nontemporal_store(val, dst);
          unsigned long long p =
              ((unsigned long long)order_key(val) << 32) | nkey;
          if (p > best[mt][v]) best[mt][v] = p;
        }
      }
    }
  }
#pragma unroll
  for (int mt = 0; mt < 2; ++mt)
#pragma unroll
    for (int v = 0; v < 8; ++v)
      atomicMax(&sbest[mt * 16 + v + roff], best[mt][v]);
  __syncthreads();
  if (threadIdx.x < 32)
    atomicMax(&argbuf[m0 + threadIdx.x], sbest[threadIdx.x]);
}

// ================================================================ launch
extern "C" void kernel_launch(void* const* d_in, const int* in_sizes, int n_in,
                              void* d_out, int out_size, void* d_ws,
                              size_t ws_size, hipStream_t stream) {
  (void)in_sizes; (void)n_in; (void)out_size; (void)ws_size;
  // inputs: images, emb_table, W_ih, W_hh, b_ih, b_hh, W_out, b_out
  const float* emb  = (const float*)d_in[1];
  const float* W_ih = (const float*)d_in[2];
  const float* W_hh = (const float*)d_in[3];
  const float* b_ih = (const float*)d_in[4];
  const float* b_hh = (const float*)d_in[5];
  const float* W_out = (const float*)d_in[6];
  const float* b_out = (const float*)d_in[7];

  float* out = (float*)d_out;                                  // logits
  float* h_f32 = out + (size_t)BATCH * MAXLEN * VOCAB;         // final h
  float* c_state = h_f32 + (size_t)BATCH * HID;                // final c (state)

  // Workspace layout (256B aligned): ~87 MB total.
  char* ws = (char*)d_ws;
  size_t off = 0;
  auto wsalloc = [&](size_t bytes) -> char* {
    char* p = ws + off;
    off = (off + bytes + 255) & ~(size_t)255;
    return p;
  };
  _Float16* Wout16 = (_Float16*)wsalloc((size_t)VOCAB * HID * 2);
  _Float16* Wih16  = (_Float16*)wsalloc((size_t)4 * HID * HID * 2);
  _Float16* Whh16  = (_Float16*)wsalloc((size_t)4 * HID * HID * 2);
  _Float16* h16a   = (_Float16*)wsalloc((size_t)BATCH * HID * 2);
  _Float16* h16b   = (_Float16*)wsalloc((size_t)BATCH * HID * 2);
  unsigned long long* arg0 = (unsigned long long*)wsalloc(BATCH * 8);
  unsigned long long* arg1 = (unsigned long long*)wsalloc(BATCH * 8);

  // One-time fp32->fp16 weight conversion; f16 W_out (77MB) then lives in L2.
  cvt_f32_to_f16<<<1024, 256, 0, stream>>>(W_out, Wout16, VOCAB * HID / 4);
  cvt_f32_to_f16<<<512, 256, 0, stream>>>(W_ih, Wih16, 4 * HID * HID / 4);
  cvt_f32_to_f16<<<512, 256, 0, stream>>>(W_hh, Whh16, 4 * HID * HID / 4);
  // arg1 acts as "previous step" buffer for t=0 (seeded with SOS).
  init_state<<<(BATCH * HID + 255) / 256, 256, 0, stream>>>(c_state, h16a,
                                                            arg1);

  for (int t = 0; t < MAXLEN; ++t) {
    const _Float16* hin = (t & 1) ? h16b : h16a;
    _Float16* hout = (t & 1) ? h16a : h16b;
    unsigned long long* acur = (t & 1) ? arg1 : arg0;   // written by step t
    unsigned long long* aprev = (t & 1) ? arg0 : arg1;  // written by step t-1
    gates_kernel<<<dim3(6, 4), 256, 0, stream>>>(
        emb, Wih16, Whh16, b_ih, b_hh, aprev, acur, hin, c_state, h_f32, hout);
    logits_kernel<<<dim3(99, 2), 256, 0, stream>>>(hout, Wout16, b_out, out, t,
                                                   acur);
  }
}
